// GNN_2353642078678
// MI455X (gfx1250) — matrix operand order repaired and checked by reading, not compiled, not run on hardware
//
#include <hip/hip_runtime.h>
#include <hip/hip_bf16.h>

#define N_NODES  100000
#define N_EDGES  1600000
#define N_GROUPS 512
#define DIN      8
#define HCH      256
#define BN_EPS   1e-5f

typedef __attribute__((ext_vector_type(16))) __bf16 v16bf;
typedef __attribute__((ext_vector_type(8)))  __bf16 v8bf;
typedef __attribute__((ext_vector_type(8)))  float  v8f;
typedef __attribute__((ext_vector_type(4)))  unsigned int v4u;
typedef __attribute__((ext_vector_type(8)))  int    v8i;
typedef __attribute__((ext_vector_type(4)))  int    v4i;

#if defined(__has_builtin)
#if __has_builtin(__builtin_amdgcn_tensor_load_to_lds) && __has_builtin(__builtin_amdgcn_s_wait_tensorcnt)
#define HAVE_TDM 1
#endif
#endif
#ifndef HAVE_TDM
#define HAVE_TDM 0
#endif

// ---------------- utility kernels ----------------

__global__ void k_fill(float* p, float v, int n) {
    int i = blockIdx.x * blockDim.x + threadIdx.x;
    if (i < n) p[i] = v;
}

__global__ void k_deg(const int* __restrict__ dst, float* __restrict__ deg, int E) {
    int e = blockIdx.x * blockDim.x + threadIdx.x;
    if (e < E) atomicAdd(&deg[dst[e]], 1.0f);
}

__global__ void k_rsqrt(float* d, int n) {
    int i = blockIdx.x * blockDim.x + threadIdx.x;
    if (i < n) d[i] = rsqrtf(d[i]);   // deg >= 1 (self loop) so always valid
}

__global__ void k_norm(const int* __restrict__ src, const int* __restrict__ dst,
                       const float* __restrict__ dis, float* __restrict__ norm,
                       int E, int N) {
    int e = blockIdx.x * blockDim.x + threadIdx.x;
    if (e >= E + N) return;
    if (e < E) norm[e] = dis[src[e]] * dis[dst[e]];
    else { float d = dis[e - E]; norm[e] = d * d; }
}

// layer 1: h = x(N x 8) @ W(8 x 256), K too small for WMMA -> plain
__global__ void k_gemm_din(const float* __restrict__ x, const float* __restrict__ W,
                           float* __restrict__ h) {
    int idx = blockIdx.x * blockDim.x + threadIdx.x;
    if (idx >= N_NODES * HCH) return;
    int r = idx >> 8, j = idx & 255;
    float s = 0.f;
#pragma unroll
    for (int k = 0; k < DIN; ++k) s += x[r * DIN + k] * W[k * HCH + j];
    h[idx] = s;
}

__global__ void k_f32_to_bf16(const float* __restrict__ in, __bf16* __restrict__ out, int n) {
    int i = blockIdx.x * blockDim.x + threadIdx.x;
    if (i < n) out[i] = (__bf16)in[i];
}

// W: [K x O] row-major f32  ->  Wt: [O x K] bf16
__global__ void k_transpose_bf16(const float* __restrict__ W, __bf16* __restrict__ Wt,
                                 int K, int O) {
    int idx = blockIdx.x * blockDim.x + threadIdx.x;
    if (idx >= K * O) return;
    int k = idx / O, o = idx - k * O;
    Wt[(size_t)o * K + k] = (__bf16)W[idx];
}

// ---------------- WMMA GEMM: C[M x Nc] = A[M x K] * Bt[Nc x K]^T (+bias)(+relu)
// block = 256 threads = 8 waves. One block per 16-row tile. The 16 x K bf16
// A-panel is staged into LDS once (TDM tensor_load_to_lds when available,
// cooperative copy otherwise); each wave then walks column tiles ct = wave+8*i,
// reading A fragments from LDS and B fragments from global (weights sit in L2).
__global__ __launch_bounds__(256)
void k_gemm_wmma(const __bf16* __restrict__ A, const __bf16* __restrict__ Bt,
                 const float* __restrict__ bias, float* __restrict__ C,
                 int M, int K, int Nc, int doRelu) {
    __shared__ __align__(16) __bf16 tileA[16 * 512];   // up to K=512 (16 KB)
    (void)M;
    int tid  = threadIdx.x;
    int wave = tid >> 5;
    int lane = tid & 31;
    int rt   = blockIdx.x;
    int half = lane >> 4;
    int l15  = lane & 15;

    const __bf16* aPanel = A + (size_t)rt * 16 * K;

#if HAVE_TDM
    if (tid < 32) {                                    // wave 0 issues the DMA
        unsigned lds_off = (unsigned)(uintptr_t)(&tileA[0]);
        unsigned long long ga = (unsigned long long)(uintptr_t)aPanel;
        v4u g0;
        g0[0] = 1u;                                    // count=1, user descriptor
        g0[1] = lds_off;                               // lds_addr [63:32]
        g0[2] = (unsigned)(ga & 0xFFFFFFFFu);          // global_addr [95:64]
        g0[3] = (unsigned)((ga >> 32) & 0x1FFFFFFu)    // global_addr [120:96]
              | (2u << 30);                            // type=2 ("image") [127:126]
        v8i g1;
        g1[0] = 0x00010000;                            // data_size=1 (2 bytes)
        g1[1] = (int)((K & 0xFFFF) << 16);             // tensor_dim0[15:0] @ [63:48]
        g1[2] = (int)(16 << 16);                       // dim0 hi=0; tensor_dim1=16 @ [95:80]
        g1[3] = (int)((K & 0xFFFF) << 16);             // dim1 hi=0; tile_dim0=K @ [127:112]
        g1[4] = 16;                                    // tile_dim1=16; tile_dim2=0
        g1[5] = K;                                     // tensor_dim0_stride[31:0]
        g1[6] = 0;                                     // stride hi; dim1_stride lo
        g1[7] = 0;
        v4i g2 = {0, 0, 0, 0};
        v4i g3 = {0, 0, 0, 0};
#if __clang_major__ >= 23
        v8i g4 = {0, 0, 0, 0, 0, 0, 0, 0};
        __builtin_amdgcn_tensor_load_to_lds(g0, g1, g2, g3, g4, 0);
#else
        __builtin_amdgcn_tensor_load_to_lds(g0, g1, g2, g3, 0);
#endif
        __builtin_amdgcn_s_wait_tensorcnt(0);
    }
#else
    for (int i = tid; i < 2 * K; i += 256)             // 16*K elems, 8 per iter
        ((v8bf*)tileA)[i] = ((const v8bf*)aPanel)[i];
#endif
    __syncthreads();

    const __bf16* arow = tileA + l15 * K;
    for (int ct = wave; ct * 16 < Nc; ct += 8) {       // wave-uniform loop
        const __bf16* brow = Bt + (size_t)(ct * 16 + l15) * K;
        v8f acc = {};
        for (int k0 = 0; k0 < K; k0 += 32) {
            // A 16x32 fragment: elems 0..7 -> K=k0+half*8+i ; 8..15 -> +16
            v8bf a_lo = *(const v8bf*)(arow + k0 + half * 8);
            v8bf a_hi = *(const v8bf*)(arow + k0 + 16 + half * 8);
            // B 32x16 fragment (pre-transposed): elems 0..7 -> K=k0+half*8+i ; 8..15 -> +16
            v8bf b_lo = *(const v8bf*)(brow + k0 + half * 8);
            v8bf b_hi = *(const v8bf*)(brow + k0 + 16 + half * 8);
            v16bf a, b;
#pragma unroll
            for (int i = 0; i < 8; ++i) {
                a[i] = a_lo[i]; a[i + 8] = a_hi[i];
                b[i] = b_lo[i]; b[i + 8] = b_hi[i];
            }
            acc = __builtin_amdgcn_wmma_f32_16x16x32_bf16(
                      false, a, false, b, (short)0, acc, false, false);
        }
        // D layout: VGPR j -> row = half*8+j, col = lane&15
#pragma unroll
        for (int j = 0; j < 8; ++j) {
            int row = rt * 16 + half * 8 + j;
            int col = ct * 16 + l15;
            float v = acc[j] + (bias ? bias[col] : 0.0f);
            if (doRelu) v = fmaxf(v, 0.0f);
            C[(size_t)row * Nc + col] = v;
        }
    }
}

// ---------------- edge aggregation: one wave per edge (incl. self loops) ----------------
__global__ void k_aggregate(const float* __restrict__ h, const int* __restrict__ src,
                            const int* __restrict__ dst, const float* __restrict__ norm,
                            float* __restrict__ agg, int E, int N) {
    long long gtid = (long long)blockIdx.x * blockDim.x + threadIdx.x;
    int wid = (int)(gtid >> 5);
    int lane = threadIdx.x & 31;
    if (wid >= E + N) return;
    int s, d;
    if (wid < E) { s = src[wid]; d = dst[wid]; }
    else         { s = wid - E; d = s; }
    float w = norm[wid];
    const float4* hs = (const float4*)(h + (size_t)s * HCH);
    float* ad = agg + (size_t)d * HCH;
    float4 a = hs[lane * 2 + 0];
    float4 b = hs[lane * 2 + 1];
    int c = lane * 8;
    atomicAdd(ad + c + 0, a.x * w); atomicAdd(ad + c + 1, a.y * w);
    atomicAdd(ad + c + 2, a.z * w); atomicAdd(ad + c + 3, a.w * w);
    atomicAdd(ad + c + 4, b.x * w); atomicAdd(ad + c + 5, b.y * w);
    atomicAdd(ad + c + 6, b.z * w); atomicAdd(ad + c + 7, b.w * w);
}

// ---------------- BatchNorm ----------------
__global__ void k_bn_stats(const float* __restrict__ agg, const float* __restrict__ bias,
                           float* __restrict__ sum, float* __restrict__ sumsq, int N) {
    int c = threadIdx.x;                 // 256 channels
    int r0 = blockIdx.x * 256;
    int r1 = r0 + 256; if (r1 > N) r1 = N;
    float bc = bias[c];
    float s = 0.f, q = 0.f;
    for (int r = r0; r < r1; ++r) {
        float v = agg[(size_t)r * HCH + c] + bc;
        s += v; q += v * v;
    }
    atomicAdd(&sum[c], s);
    atomicAdd(&sumsq[c], q);
}

__global__ void k_bn_relu(const float* __restrict__ agg, const float* __restrict__ bias,
                          const float* __restrict__ sum, const float* __restrict__ sumsq,
                          const float* __restrict__ gam, const float* __restrict__ bet,
                          float* __restrict__ out, int N) {
    int idx = blockIdx.x * blockDim.x + threadIdx.x;
    if (idx >= N * HCH) return;
    int c = idx & 255;
    float invN = 1.0f / (float)N;
    float mu  = sum[c] * invN;
    float var = sumsq[c] * invN - mu * mu;
    float v = (agg[idx] + bias[c] - mu) * rsqrtf(var + BN_EPS) * gam[c] + bet[c];
    out[idx] = fmaxf(v, 0.0f);
}

// ---------------- pooling ----------------
__global__ void k_count(const int* __restrict__ batch, float* __restrict__ cnt, int N) {
    int n = blockIdx.x * blockDim.x + threadIdx.x;
    if (n < N) atomicAdd(&cnt[batch[n]], 1.0f);
}

__global__ void k_pool(const float* __restrict__ act, const int* __restrict__ batch,
                       float* __restrict__ psum, int* __restrict__ pmax, int N) {
    int idx = blockIdx.x * blockDim.x + threadIdx.x;
    if (idx >= N * HCH) return;
    int n = idx >> 8, c = idx & 255;
    int g = batch[n];
    float v = act[idx];
    atomicAdd(&psum[g * HCH + c], v);
    atomicMax(&pmax[g * HCH + c], __float_as_int(v));   // v >= 0 post-ReLU -> bits monotonic
}

__global__ void k_pool_fin(const float* __restrict__ psum, const int* __restrict__ pmax,
                           const float* __restrict__ cnt, float* __restrict__ p) {
    int idx = blockIdx.x * blockDim.x + threadIdx.x;
    if (idx >= N_GROUPS * HCH) return;
    int g = idx >> 8, c = idx & 255;
    p[g * 2 * HCH + c]       = psum[idx] / fmaxf(cnt[g], 1.0f);
    p[g * 2 * HCH + HCH + c] = __int_as_float(pmax[idx]);
}

// ---------------- final 64 -> 1 projection ----------------
__global__ void k_fc3(const float* __restrict__ q, const float* __restrict__ W,
                      const float* __restrict__ b, float* __restrict__ out) {
    int g = blockIdx.x * blockDim.x + threadIdx.x;
    if (g >= N_GROUPS) return;
    float s = b[0];
#pragma unroll
    for (int k = 0; k < 64; ++k) s += q[g * 64 + k] * W[k];
    out[g] = s;
}

// ---------------- host orchestration ----------------
extern "C" void kernel_launch(void* const* d_in, const int* in_sizes, int n_in,
                              void* d_out, int out_size, void* d_ws, size_t ws_size,
                              hipStream_t stream) {
    (void)in_sizes; (void)n_in; (void)out_size; (void)ws_size;

    const float* x     = (const float*)d_in[0];
    const int*   srcI  = (const int*)d_in[1];
    const int*   dstI  = srcI + N_EDGES;
    const int*   batch = (const int*)d_in[2];
    const float* W[4]  = {(const float*)d_in[3], (const float*)d_in[5],
                          (const float*)d_in[7], (const float*)d_in[9]};
    const float* bs[4] = {(const float*)d_in[4], (const float*)d_in[6],
                          (const float*)d_in[8], (const float*)d_in[10]};
    const float* ga[4] = {(const float*)d_in[11], (const float*)d_in[13],
                          (const float*)d_in[15], (const float*)d_in[17]};
    const float* bt[4] = {(const float*)d_in[12], (const float*)d_in[14],
                          (const float*)d_in[16], (const float*)d_in[18]};
    const float* fW1 = (const float*)d_in[19]; const float* fb1 = (const float*)d_in[20];
    const float* fW2 = (const float*)d_in[21]; const float* fb2 = (const float*)d_in[22];
    const float* fW3 = (const float*)d_in[23]; const float* fb3 = (const float*)d_in[24];

    char* ws = (char*)d_ws;
    size_t off = 0;
    auto carve = [&](size_t bytes) -> void* {
        void* p = ws + off;
        off += (bytes + 255) & ~(size_t)255;
        return p;
    };
    float*  B1    = (float*) carve((size_t)N_NODES * HCH * 4);   // act / gemm-out
    float*  B2    = (float*) carve((size_t)N_NODES * HCH * 4);   // aggregation
    __bf16* Bbf   = (__bf16*)carve((size_t)N_NODES * HCH * 2);   // bf16 staging (A)
    float*  norm  = (float*) carve((size_t)(N_EDGES + N_NODES) * 4);
    float*  dis   = (float*) carve((size_t)N_NODES * 4);         // deg -> rsqrt(deg)
    __bf16* Wt    = (__bf16*)carve((size_t)512 * 256 * 2);       // transposed bf16 weights
    float*  sum   = (float*) carve(HCH * 4);
    float*  sumsq = (float*) carve(HCH * 4);
    float*  psum  = (float*) carve((size_t)N_GROUPS * HCH * 4);
    int*    pmax  = (int*)   carve((size_t)N_GROUPS * HCH * 4);
    float*  cnt   = (float*) carve((size_t)N_GROUPS * 4);
    float*  pcat  = (float*) carve((size_t)N_GROUPS * 2 * HCH * 4);
    float*  q1    = (float*) carve((size_t)N_GROUPS * 256 * 4);
    float*  q2    = (float*) carve((size_t)N_GROUPS * 64 * 4);

    const int TB = 256;
    const int elemNH = N_NODES * HCH;
    const int gridNH = (elemNH + TB - 1) / TB;

    // degrees (self loop -> init 1), dis = rsqrt(deg), per-edge norm
    k_fill<<<(N_NODES + TB - 1) / TB, TB, 0, stream>>>(dis, 1.0f, N_NODES);
    k_deg<<<(N_EDGES + TB - 1) / TB, TB, 0, stream>>>(dstI, dis, N_EDGES);
    k_rsqrt<<<(N_NODES + TB - 1) / TB, TB, 0, stream>>>(dis, N_NODES);
    k_norm<<<(N_EDGES + N_NODES + TB - 1) / TB, TB, 0, stream>>>(srcI, dstI, dis, norm,
                                                                 N_EDGES, N_NODES);

    long long aggThreads = (long long)(N_EDGES + N_NODES) * 32;
    int aggGrid = (int)((aggThreads + TB - 1) / TB);

    for (int l = 0; l < 4; ++l) {
        if (l == 0) {
            k_gemm_din<<<gridNH, TB, 0, stream>>>(x, W[0], B1);
        } else {
            k_f32_to_bf16<<<gridNH, TB, 0, stream>>>(B1, Bbf, elemNH);
            k_transpose_bf16<<<(HCH * HCH + TB - 1) / TB, TB, 0, stream>>>(W[l], Wt, HCH, HCH);
            k_gemm_wmma<<<dim3(N_NODES / 16), TB, 0, stream>>>(Bbf, Wt, nullptr, B1,
                                                               N_NODES, HCH, HCH, 0);
        }
        hipMemsetAsync(B2, 0, (size_t)elemNH * 4, stream);
        k_aggregate<<<aggGrid, TB, 0, stream>>>(B1, srcI, dstI, norm, B2,
                                                N_EDGES, N_NODES);
        hipMemsetAsync(sum, 0, HCH * 4, stream);
        hipMemsetAsync(sumsq, 0, HCH * 4, stream);
        k_bn_stats<<<(N_NODES + 255) / 256, HCH, 0, stream>>>(B2, bs[l], sum, sumsq, N_NODES);
        k_bn_relu<<<gridNH, TB, 0, stream>>>(B2, bs[l], sum, sumsq, ga[l], bt[l],
                                             B1, N_NODES);
    }

    // pooling
    hipMemsetAsync(psum, 0, (size_t)N_GROUPS * HCH * 4, stream);
    hipMemsetAsync(pmax, 0, (size_t)N_GROUPS * HCH * 4, stream);
    hipMemsetAsync(cnt, 0, (size_t)N_GROUPS * 4, stream);
    k_count<<<(N_NODES + TB - 1) / TB, TB, 0, stream>>>(batch, cnt, N_NODES);
    k_pool<<<gridNH, TB, 0, stream>>>(B1, batch, psum, pmax, N_NODES);
    k_pool_fin<<<(N_GROUPS * HCH + TB - 1) / TB, TB, 0, stream>>>(psum, pmax, cnt, pcat);

    // fc1: pcat[512 x 512] @ fW1[512 x 256] + fb1, ReLU   (WMMA)
    k_f32_to_bf16<<<(N_GROUPS * 512 + TB - 1) / TB, TB, 0, stream>>>(pcat, Bbf, N_GROUPS * 512);
    k_transpose_bf16<<<(512 * 256 + TB - 1) / TB, TB, 0, stream>>>(fW1, Wt, 512, 256);
    k_gemm_wmma<<<dim3(N_GROUPS / 16), TB, 0, stream>>>(Bbf, Wt, fb1, q1,
                                                        N_GROUPS, 512, 256, 1);
    // fc2: q1[512 x 256] @ fW2[256 x 64] + fb2, ReLU      (WMMA)
    k_f32_to_bf16<<<(N_GROUPS * 256 + TB - 1) / TB, TB, 0, stream>>>(q1, Bbf, N_GROUPS * 256);
    k_transpose_bf16<<<(256 * 64 + TB - 1) / TB, TB, 0, stream>>>(fW2, Wt, 256, 64);
    k_gemm_wmma<<<dim3(N_GROUPS / 16), TB, 0, stream>>>(Bbf, Wt, fb2, q2,
                                                        N_GROUPS, 256, 64, 1);
    // fc3: q2[512 x 64] @ fW3[64 x 1] + fb3
    k_fc3<<<(N_GROUPS + TB - 1) / TB, TB, 0, stream>>>(q2, fW3, fb3, (float*)d_out);
}
